// CustomCausalAttention_53592601919680
// MI455X (gfx1250) — matrix operand order repaired and checked by reading, not compiled, not run on hardware
//
#include <hip/hip_runtime.h>

typedef _Float16 v16h __attribute__((ext_vector_type(16)));
typedef float    v8f  __attribute__((ext_vector_type(8)));

#define WMMA_F16(A, B, C) \
  __builtin_amdgcn_wmma_f32_16x16x32_f16(false, (A), false, (B), (short)0, (C), false, false)

namespace {
constexpr int   Bc = 2, Hc = 16, Nc = 2048, Dc = 64;
// logits in base-2 domain: lg2 = (2*S - |q|^2 - |k|^2) * (1/sqrt(D)) * log2(e), clamped <= 0
constexpr float SC2     = 0.125f * 1.4426950408889634f;  // scale * log2(e)
constexpr float TWO_SC2 = 2.0f * SC2;
}

union HU  { v16h h; unsigned int u[8]; };
union H4U { _Float16 h[4]; uint2 u2; };

// Async copy: VDST vgpr = LDS byte address, VADDR vgpr = 32-bit offset,
// SADDR = 64-bit base; the instruction offset is added to BOTH sides.
#define ASYNC_CP(ldst, off, gbase, IMM)                                        \
  asm volatile("global_load_async_to_lds_b128 %0, %1, %2 offset:" IMM          \
               :: "v"(ldst), "v"(off),                                         \
                  "s"((unsigned long long)(uintptr_t)(gbase)) : "memory")
#define ASYNC_TILE8K(ldst, off, gbase) do {                                    \
    ASYNC_CP(ldst, off, gbase, "0");                                           \
    ASYNC_CP(ldst, off, gbase, "2048");                                        \
    ASYNC_CP(ldst, off, gbase, "4096");                                        \
    ASYNC_CP(ldst, off, gbase, "6144"); } while (0)

// ---------------------------------------------------------------------------
// Pre-pass: fp32 -> f16 for Q,K (row-major), V transposed per 64-key tile,
// plus qs_n = -|q|^2*SC2 and ks_n = -|k|^2*SC2.  Pure memory-bound, done once.
// ---------------------------------------------------------------------------
__global__ __launch_bounds__(128) void rbf_attn_prepass(
    const float* __restrict__ q, const float* __restrict__ k,
    const float* __restrict__ v, _Float16* __restrict__ qh,
    _Float16* __restrict__ kh, _Float16* __restrict__ vth,
    float* __restrict__ qsn, float* __restrict__ ksn)
{
  __shared__ __align__(16) _Float16 Vt[64 * 64];
  const int tid = threadIdx.x;
  const int t0  = blockIdx.x;          // 64-row tile index
  const int bh  = blockIdx.y;
  const size_t fbase = ((size_t)bh * Nc + (size_t)t0 * 64) * Dc;   // element idx
  const float4* q4 = reinterpret_cast<const float4*>(q + fbase);
  const float4* k4 = reinterpret_cast<const float4*>(k + fbase);
  const float4* v4 = reinterpret_cast<const float4*>(v + fbase);
  _Float16* qhb = qh + fbase;
  _Float16* khb = kh + fbase;

  #pragma unroll
  for (int j = 0; j < 8; ++j) {
    const int idx = tid + j * 128;     // 1024 float4 per tile
    float4 a = q4[idx];
    H4U ta; ta.h[0]=(_Float16)a.x; ta.h[1]=(_Float16)a.y;
            ta.h[2]=(_Float16)a.z; ta.h[3]=(_Float16)a.w;
    *reinterpret_cast<uint2*>(qhb + idx * 4) = ta.u2;
    float4 b = k4[idx];
    H4U tb; tb.h[0]=(_Float16)b.x; tb.h[1]=(_Float16)b.y;
            tb.h[2]=(_Float16)b.z; tb.h[3]=(_Float16)b.w;
    *reinterpret_cast<uint2*>(khb + idx * 4) = tb.u2;
    float4 c = v4[idx];
    const int key = idx >> 4, col = idx & 15;
    Vt[(col * 4 + 0) * 64 + key] = (_Float16)c.x;
    Vt[(col * 4 + 1) * 64 + key] = (_Float16)c.y;
    Vt[(col * 4 + 2) * 64 + key] = (_Float16)c.z;
    Vt[(col * 4 + 3) * 64 + key] = (_Float16)c.w;
  }
  if (tid < 64) {                      // fp32 row sums of squares
    const float* qr = q + fbase + (size_t)tid * Dc;
    const float* kr = k + fbase + (size_t)tid * Dc;
    float sq = 0.f, sk = 0.f;
    #pragma unroll 8
    for (int i = 0; i < Dc; ++i) { sq += qr[i]*qr[i]; sk += kr[i]*kr[i]; }
    qsn[(size_t)bh * Nc + t0 * 64 + tid] = -sq * SC2;
    ksn[(size_t)bh * Nc + t0 * 64 + tid] = -sk * SC2;
  }
  __syncthreads();
  // write transposed V tile, coalesced b128
  const uint4* vl = reinterpret_cast<const uint4*>(Vt);
  uint4* vg = reinterpret_cast<uint4*>(vth + ((size_t)bh * 32 + t0) * 4096);
  #pragma unroll
  for (int j = 0; j < 4; ++j) vg[tid + j * 128] = vl[tid + j * 128];
}

// ---------------------------------------------------------------------------
// Main flash-attention kernel: f16 WMMA, async double-buffered K/V staging.
// ---------------------------------------------------------------------------
__global__ __launch_bounds__(128) void rbf_attn_main(
    const _Float16* __restrict__ qh, const _Float16* __restrict__ kh,
    const _Float16* __restrict__ vth, const float* __restrict__ qsn,
    const float* __restrict__ ksn, float* __restrict__ out)
{
  // Double-buffered tiles: [buf][0]=K row-major [key64][d64], [buf][1]=V^T [d64][key64]
  __shared__ __align__(16) _Float16 KV[2][2][64 * 64];
  __shared__ __align__(16) _Float16 Plds[4][16 * 64];

  const int tid  = threadIdx.x;
  const int wid  = tid >> 5;
  const int lane = tid & 31;
  const int l16  = lane & 15;
  const int g    = lane >> 4;

  const int bh = blockIdx.y;
  const int q0 = blockIdx.x * 64;
  const int mbase = q0 + wid * 16;
  float* ob = out + (size_t)bh * Nc * Dc;

  // ---- Q A-fragments straight from f16 (A layout: row M=l16; VGPR r holds
  //      K = (r/4)*16 + g*8 + (r%4)*2, +1) ----
  const _Float16* qrow = qh + ((size_t)bh * Nc + mbase + l16) * Dc;
  HU qa0, qa1;
  #pragma unroll
  for (int r = 0; r < 8; ++r) {
    const int d0 = ((r >> 2) * 16) + g * 8 + (r & 3) * 2;
    qa0.u[r] = *reinterpret_cast<const unsigned int*>(qrow + d0);
    qa1.u[r] = *reinterpret_cast<const unsigned int*>(qrow + d0 + 32);
  }
  float qs_n[8];
  #pragma unroll
  for (int r = 0; r < 8; ++r)
    qs_n[r] = qsn[(size_t)bh * Nc + mbase + r + 8 * g];

  v8f  o0 = {}, o1 = {}, o2 = {}, o3 = {};
  float lacc[8];
  #pragma unroll
  for (int r = 0; r < 8; ++r) lacc[r] = 0.f;

  // ---- async staging setup ----
  const unsigned voff  = (unsigned)tid * 16u;            // per-thread 16B slice
  const unsigned ldsB0 = (unsigned)(uintptr_t)&KV[0][0][0] + voff;
  const unsigned ldsB1 = (unsigned)(uintptr_t)&KV[1][0][0] + voff;
  const _Float16* kbh = kh  + (size_t)bh * Nc * Dc;
  const _Float16* vbh = vth + (size_t)bh * 32 * 4096;

  const int kb_count = blockIdx.x + 1;
  // prologue: tile 0 -> buffer 0 (8 async b128 per thread = K 8KB + V^T 8KB)
  ASYNC_TILE8K(ldsB0,        voff, kbh);
  ASYNC_TILE8K(ldsB0 + 8192, voff, vbh);

  for (int kb = 0; kb < kb_count; ++kb) {
    __syncthreads();                       // all waves done with buf[(kb+1)&1]
    if (kb + 1 < kb_count) {               // prefetch next tile into other buf
      const unsigned ldsd = ((kb + 1) & 1) ? ldsB1 : ldsB0;
      ASYNC_TILE8K(ldsd,        voff, kbh + (size_t)(kb + 1) * 64 * Dc);
      ASYNC_TILE8K(ldsd + 8192, voff, vbh + (size_t)(kb + 1) * 4096);
      asm volatile("s_wait_asynccnt 0x8" ::: "memory");  // tile kb done, kb+1 in flight
    } else {
      asm volatile("s_wait_asynccnt 0x0" ::: "memory");
    }
    __syncthreads();                       // tile kb visible to all waves

    const _Float16* Kl = KV[kb & 1][0];
    const _Float16* Vl = KV[kb & 1][1];

    // ---------------- S = Q * K^T (four 16x16 key groups) ----------------
    v8f c[4];
    #pragma unroll
    for (int gi = 0; gi < 4; ++gi) {
      HU blo, bhi;
      const int key = gi * 16 + l16;
      #pragma unroll
      for (int r = 0; r < 8; ++r) {
        const int d = ((r >> 2) * 16) + g * 8 + (r & 3) * 2;
        blo.u[r] = *reinterpret_cast<const unsigned int*>(&Kl[key * 64 + d]);
        bhi.u[r] = *reinterpret_cast<const unsigned int*>(&Kl[key * 64 + d + 32]);
      }
      v8f acc = {};
      acc = WMMA_F16(qa0.h, blo.h, acc);
      acc = WMMA_F16(qa1.h, bhi.h, acc);
      c[gi] = acc;
    }

    // ---------------- probabilities (row max is exactly 0) ----------------
    float ks_n[4];
    #pragma unroll
    for (int gi = 0; gi < 4; ++gi)
      ks_n[gi] = ksn[(size_t)bh * Nc + kb * 64 + gi * 16 + l16];

    float p[4][8];
    const bool edge = (kb * 64 + 63) > mbase;   // wave-uniform
    if (edge) {
      #pragma unroll
      for (int gi = 0; gi < 4; ++gi) {
        const int n = kb * 64 + gi * 16 + l16;
        #pragma unroll
        for (int r = 0; r < 8; ++r) {
          const int mg = mbase + r + 8 * g;
          float lg = fminf(fmaf(c[gi][r], TWO_SC2, qs_n[r] + ks_n[gi]), 0.f);
          lg = (n <= mg) ? lg : -__builtin_inff();
          const float e = __builtin_amdgcn_exp2f(lg);
          lacc[r] += e;
          p[gi][r] = e;
        }
      }
    } else {
      #pragma unroll
      for (int gi = 0; gi < 4; ++gi) {
        #pragma unroll
        for (int r = 0; r < 8; ++r) {
          const float lg = fminf(fmaf(c[gi][r], TWO_SC2, qs_n[r] + ks_n[gi]), 0.f);
          const float e = __builtin_amdgcn_exp2f(lg);
          lacc[r] += e;
          p[gi][r] = e;
        }
      }
    }

    // ---------------- P: C layout -> A layout via per-wave LDS -------------
    _Float16* pl = Plds[wid];
    #pragma unroll
    for (int gi = 0; gi < 4; ++gi)
      #pragma unroll
      for (int r = 0; r < 8; ++r)
        pl[(r + 8 * g) * 64 + gi * 16 + l16] = (_Float16)p[gi][r];
    asm volatile("s_wait_dscnt 0x0" ::: "memory");
    HU pa0, pa1;
    #pragma unroll
    for (int r = 0; r < 8; ++r) {
      const int kk = ((r >> 2) * 16) + g * 8 + (r & 3) * 2;
      pa0.u[r] = *reinterpret_cast<const unsigned int*>(&pl[l16 * 64 + kk]);
      pa1.u[r] = *reinterpret_cast<const unsigned int*>(&pl[l16 * 64 + kk + 32]);
    }

    // ---------------- O += P * V ------------------------------------------
    #pragma unroll
    for (int dg = 0; dg < 4; ++dg) {
      HU vlo, vhi;
      const int d = dg * 16 + l16;
      #pragma unroll
      for (int r = 0; r < 8; ++r) {
        const int key = ((r >> 2) * 16) + g * 8 + (r & 3) * 2;
        vlo.u[r] = *reinterpret_cast<const unsigned int*>(&Vl[d * 64 + key]);
        vhi.u[r] = *reinterpret_cast<const unsigned int*>(&Vl[d * 64 + key + 32]);
      }
      v8f acc = (dg == 0) ? o0 : (dg == 1) ? o1 : (dg == 2) ? o2 : o3;
      acc = WMMA_F16(pa0.h, vlo.h, acc);
      acc = WMMA_F16(pa1.h, vhi.h, acc);
      if (dg == 0) o0 = acc; else if (dg == 1) o1 = acc;
      else if (dg == 2) o2 = acc; else o3 = acc;
    }
  }

  // ---------------- final row-sum reduction + normalize + writeback --------
  #pragma unroll
  for (int r = 0; r < 8; ++r) {
    float s = lacc[r];
    s += __shfl_xor(s, 1); s += __shfl_xor(s, 2);
    s += __shfl_xor(s, 4); s += __shfl_xor(s, 8);
    const float inv = 1.0f / s;
    const int mg = mbase + r + 8 * g;
    float* orow = ob + (size_t)mg * Dc;
    orow[0 * 16 + l16] = o0[r] * inv;
    orow[1 * 16 + l16] = o1[r] * inv;
    orow[2 * 16 + l16] = o2[r] * inv;
    orow[3 * 16 + l16] = o3[r] * inv;
  }
}

extern "C" void kernel_launch(void* const* d_in, const int* in_sizes, int n_in,
                              void* d_out, int out_size, void* d_ws, size_t ws_size,
                              hipStream_t stream) {
  (void)in_sizes; (void)n_in; (void)out_size; (void)ws_size;
  const float* q = (const float*)d_in[0];
  const float* k = (const float*)d_in[1];
  const float* v = (const float*)d_in[2];
  float* out = (float*)d_out;

  // Workspace layout (24.5 MB): Qh | Kh | VtT (f16, 8 MB each) | qs_n | ks_n
  const size_t nElem = (size_t)Bc * Hc * Nc * Dc;
  _Float16* qhp = (_Float16*)d_ws;
  _Float16* khp = qhp + nElem;
  _Float16* vtp = khp + nElem;
  float*    qsn = (float*)(vtp + nElem);
  float*    ksn = qsn + (size_t)Bc * Hc * Nc;

  dim3 grid(Nc / 64, Bc * Hc);   // (32, 32)
  rbf_attn_prepass<<<grid, 128, 0, stream>>>(q, k, v, qhp, khp, vtp, qsn, ksn);
  rbf_attn_main<<<grid, 128, 0, stream>>>(qhp, khp, vtp, qsn, ksn, out);
}
